// get_model_10909216931866
// MI455X (gfx1250) — matrix operand order repaired and checked by reading, not compiled, not run on hardware
//
#include <hip/hip_runtime.h>

// ---------------------------------------------------------------------------
// PointNet++ MSG (2-pass) forward for MI455X / gfx1250.
// - All cbr (conv+BN+ReLU) layers run through a WMMA f16 GEMM
//   (v_wmma_f32_16x16x32_f16), f32 accumulate, BN folded into scale/shift.
// - Activations and weights are stored in WMMA *fragment* layout so the GEMM
//   inner loop is pure 32B vector loads + v_wmma (no per-element addressing).
// - Every layer has Mtiles % 4 == 0, so each wave owns exactly 4 output tiles
//   with straight-line (EXEC=all-ones) code: no divergence around WMMA.
// - Geometry (FPS / ball query / 3-NN) computed ONCE and shared by both
//   passes (coordinates are identical in pass 1 and pass 2).
// ---------------------------------------------------------------------------

typedef _Float16 v16h __attribute__((ext_vector_type(16)));
typedef float    v8f  __attribute__((ext_vector_type(8)));

static inline int ceil_div(int a, int b) { return (a + b - 1) / b; }
static inline int pad32i(int x) { return (x + 31) & ~31; }

#define GTID (blockIdx.x * blockDim.x + threadIdx.x)

// ---------------------------------------------------------------------------
// WMMA B-fragment packed activation layout (wave32, 16x16x32 f16):
//   element (channel c, column m)  ->  tile nt=m/16, k-block kblk=c/32,
//   lane = ((c%32)/8)%2 * 16 + m%16, half j = c%8 + (c%32)/16*8.
// Buffer is [Mtiles][Cip/32][32 lanes][16 halves] = M*Cip halves total.
// ---------------------------------------------------------------------------
__device__ __forceinline__ size_t xidx(int c, int m, int Cip) {
  int nt = m >> 4, l16 = m & 15;
  int kblk = c >> 5, cr = c & 31;
  int h = (cr >> 3) & 1, j = (cr & 7) | ((cr >> 4) << 3);
  int nkb = Cip >> 5;
  return ((((size_t)nt * nkb + kblk) * 32) + (h * 16 + l16)) * 16 + j;
}

// ---------------------------------------------------------------------------
// layout helpers: f32 feature maps are "CM": F[c * (B*n) + b*n + i]
// ---------------------------------------------------------------------------
__global__ void k_bmaj2cm(const float* __restrict__ src, int Bb, int C, int N,
                          float* __restrict__ dst) {
  int BN = Bb * N;
  int id = GTID;
  if (id >= C * BN) return;
  int c = id / BN, m = id - c * BN;
  int b = m / N, n = m - b * N;
  dst[id] = src[((size_t)b * C + c) * N + n];
}

// ---------------------------------------------------------------------------
// Farthest point sampling: one block per batch, sequential argmax chain.
// ---------------------------------------------------------------------------
__global__ void k_fps(const float* __restrict__ coords, int colStride, int Ns,
                      int Np, float* __restrict__ dist, int* __restrict__ fidx) {
  __shared__ float sd[256];
  __shared__ int   si[256];
  __shared__ int   sfar;
  int b = blockIdx.x, t = threadIdx.x, bs = blockDim.x;
  const float* X = coords + b * Ns;
  float* D = dist + b * Ns;
  for (int n = t; n < Ns; n += bs) D[n] = 1e10f;
  if (t == 0) sfar = 0;
  __syncthreads();
  for (int it = 0; it < Np; ++it) {
    int far = sfar;
    if (t == 0) fidx[b * Np + it] = far;
    float cx = X[far], cy = X[colStride + far], cz = X[2 * colStride + far];
    float bd = -1.f; int bi = 0;
    for (int n = t; n < Ns; n += bs) {
      float dx = X[n] - cx, dy = X[colStride + n] - cy, dz = X[2 * colStride + n] - cz;
      float d2 = dx * dx + dy * dy + dz * dz;
      float d = D[n];
      d = d2 < d ? d2 : d;
      D[n] = d;
      if (d > bd || (d == bd && n < bi)) { bd = d; bi = n; }
    }
    sd[t] = bd; si[t] = bi;
    __syncthreads();
    for (int s = bs >> 1; s > 0; s >>= 1) {
      if (t < s) {
        if (sd[t + s] > sd[t] || (sd[t + s] == sd[t] && si[t + s] < si[t])) {
          sd[t] = sd[t + s]; si[t] = si[t + s];
        }
      }
      __syncthreads();
    }
    if (t == 0) sfar = si[0];
    __syncthreads();
  }
}

__global__ void k_centers(const float* __restrict__ src, int srcStride,
                          const int* __restrict__ fidx, int Ns, int Np, int Bb,
                          float* __restrict__ ctr) {
  int BNq = Bb * Np;
  int q = GTID;
  if (q >= BNq) return;
  int b = q / Np;
  int g = b * Ns + fidx[q];
  for (int c = 0; c < 3; ++c) ctr[c * BNq + q] = src[(size_t)c * srcStride + g];
}

// ---------------------------------------------------------------------------
// Ball query: first K in-range indices in ascending order (matches reference
// sort-then-truncate), padded with the first hit (center always qualifies).
// ---------------------------------------------------------------------------
__global__ void k_ball(const float* __restrict__ src, int srcStride, int Ns,
                       const float* __restrict__ ctr, int Np, int Bb, float r2,
                       int K, int* __restrict__ gidx) {
  int BNq = Bb * Np;
  int q = GTID;
  if (q >= BNq) return;
  int b = q / Np;
  float cx = ctr[q], cy = ctr[BNq + q], cz = ctr[2 * BNq + q];
  const float* X = src + b * Ns;
  int cnt = 0;
  for (int n = 0; n < Ns && cnt < K; ++n) {
    float dx = X[n] - cx, dy = X[srcStride + n] - cy, dz = X[2 * srcStride + n] - cz;
    if (dx * dx + dy * dy + dz * dz <= r2) gidx[q * K + cnt++] = n;
  }
  int f = cnt > 0 ? gidx[q * K] : 0;
  for (; cnt < K; ++cnt) gidx[q * K + cnt] = f;
}

// ---------------------------------------------------------------------------
// Grouping: packed X, m = (b*Np+i)*K + k; channels = [features, dxyz, 0pad]
// ---------------------------------------------------------------------------
__global__ void k_group(const float* __restrict__ F, int Cf, int fStride,
                        const float* __restrict__ srcc, int srcStride,
                        const float* __restrict__ ctr, const int* __restrict__ gidx,
                        int Ns, int Np, int K, _Float16* __restrict__ X, int Cip,
                        int M, int BNq) {
  int m = GTID;
  if (m >= M) return;
  int q = m / K;
  int b = q / Np;
  int gs = b * Ns + gidx[m];
  for (int c = 0; c < Cf; ++c)
    X[xidx(c, m, Cip)] = (_Float16)F[(size_t)c * fStride + gs];
  for (int c = 0; c < 3; ++c)
    X[xidx(Cf + c, m, Cip)] =
        (_Float16)(srcc[(size_t)c * srcStride + gs] - ctr[(size_t)c * BNq + q]);
  for (int c = Cf + 3; c < Cip; ++c) X[xidx(c, m, Cip)] = (_Float16)0.f;
}

// ---------------------------------------------------------------------------
// Pack conv+BN: f32 weights -> zero-padded f16 in WMMA A-fragment layout:
//   lane = h*16 + (row%16); halves j<8 -> K = kblk*32 + h*8+j, j>=8 -> +16.
// BN folded: y = relu(acc*s[m] + t[m]); padded rows get s=t=0.
// ---------------------------------------------------------------------------
__global__ void k_pack(const float* __restrict__ w, const float* __restrict__ bia,
                       const float* __restrict__ g, const float* __restrict__ bt,
                       const float* __restrict__ mn, const float* __restrict__ vr,
                       int Cout, int Cin, int Cop, int Cip,
                       _Float16* __restrict__ Wh, float* __restrict__ s,
                       float* __restrict__ t) {
  int id = GTID;
  if (id < Cop * Cip) {
    int j = id & 15;
    int lane = (id >> 4) & 31;
    int blk = id >> 9;             // ct * (Cip/32) + kblk
    int nkb = Cip >> 5;
    int ct = blk / nkb, kblk = blk - ct * nkb;
    int l16 = lane & 15, h = lane >> 4;
    int r = ct * 16 + l16;
    int kin = (j < 8) ? (h * 8 + j) : (16 + h * 8 + (j - 8));
    int c = kblk * 32 + kin;
    Wh[id] = (r < Cout && c < Cin) ? (_Float16)w[(size_t)r * Cin + c] : (_Float16)0.f;
  }
  if (id < Cop) {
    if (id < Cout) {
      float sv = g[id] * rsqrtf(vr[id] + 1e-5f);
      s[id] = sv;
      t[id] = (bia[id] - mn[id]) * sv + bt[id];
    } else {
      s[id] = 0.f; t[id] = 0.f;
    }
  }
}

// ---------------------------------------------------------------------------
// WMMA GEMM: Y[m,n] = relu(s[m]*(sum_k Wh[m,k]*X[k,n]) + t[m]).
// Wave owns exactly 4 adjacent 16x16 output tiles (Mtiles % 4 == 0 for every
// layer in this model). Straight-line inner loop: 1 A load + 4 B loads +
// 4 v_wmma, running pointers (no per-iteration address rebuilds).
// ---------------------------------------------------------------------------
__global__ void k_wmma_gemm(const _Float16* __restrict__ Wh,
                            const float* __restrict__ s, const float* __restrict__ t,
                            const _Float16* __restrict__ X,
                            float* __restrict__ Yf, _Float16* __restrict__ Yh,
                            int M, int Cip, int Cop, int Mtiles) {
  int wid = threadIdx.x >> 5, lane = threadIdx.x & 31;
  int g4 = blockIdx.y * 4 + wid;        // group of 4 m-tiles
  int mt0 = g4 * 4;
  if (mt0 >= Mtiles) return;            // wave-uniform exit
  int nkb = Cip >> 5;
  int ctile = blockIdx.x;

  v8f acc0, acc1, acc2, acc3;
#pragma unroll
  for (int r = 0; r < 8; ++r) { acc0[r] = 0.f; acc1[r] = 0.f; acc2[r] = 0.f; acc3[r] = 0.f; }

  size_t tstep = (size_t)nkb * 32;      // v16h elements per m-tile
  const v16h* ap = (const v16h*)Wh + (size_t)ctile * tstep + lane;
  const v16h* b0 = (const v16h*)X + (size_t)mt0 * tstep + lane;
  const v16h* b1 = b0 + tstep;
  const v16h* b2 = b1 + tstep;
  const v16h* b3 = b2 + tstep;

  for (int kblk = 0; kblk < nkb; ++kblk) {
    v16h a  = *ap;
    v16h m0 = *b0;
    v16h m1 = *b1;
    v16h m2 = *b2;
    v16h m3 = *b3;
    if (kblk + 1 < nkb) __builtin_prefetch((const void*)(ap + 32), 0, 1);
    acc0 = __builtin_amdgcn_wmma_f32_16x16x32_f16(false, a, false, m0, (short)0, acc0, false, false);
    acc1 = __builtin_amdgcn_wmma_f32_16x16x32_f16(false, a, false, m1, (short)0, acc1, false, false);
    acc2 = __builtin_amdgcn_wmma_f32_16x16x32_f16(false, a, false, m2, (short)0, acc2, false, false);
    acc3 = __builtin_amdgcn_wmma_f32_16x16x32_f16(false, a, false, m3, (short)0, acc3, false, false);
    ap += 32; b0 += 32; b1 += 32; b2 += 32; b3 += 32;
  }

  int h = lane >> 4, l16 = lane & 15;
  float sv[8], tv[8];
#pragma unroll
  for (int r = 0; r < 8; ++r) {
    int mm = ctile * 16 + h * 8 + r;    // D VGPR r: row = h*8 + r
    sv[r] = s[mm]; tv[r] = t[mm];
  }
#pragma unroll
  for (int u = 0; u < 4; ++u) {
    const v8f& a = u == 0 ? acc0 : u == 1 ? acc1 : u == 2 ? acc2 : acc3;
    int n = (mt0 + u) * 16 + l16;
    if (n < M) {
#pragma unroll
      for (int r = 0; r < 8; ++r) {
        int mm = ctile * 16 + h * 8 + r;
        float v = a[r] * sv[r] + tv[r];
        v = v > 0.f ? v : 0.f;
        Yf[(size_t)mm * M + n] = v;
        if (Yh) Yh[xidx(mm, n, Cop)] = (_Float16)v;  // feeds next GEMM's B
      }
    }
  }
}

// max over K neighbors -> dst[(rowOff+c)*BNq + q]
__global__ void k_maxpool(const float* __restrict__ Y, int M, int Cout, int K,
                          int BNq, float* __restrict__ dst, int rowOff) {
  int id = GTID;
  if (id >= BNq * Cout) return;
  int c = id / BNq, q = id - c * BNq;
  const float* p = Y + (size_t)c * M + (size_t)q * K;
  float bv = p[0];
  for (int k = 1; k < K; ++k) bv = fmaxf(bv, p[k]);
  dst[(size_t)(rowOff + c) * BNq + q] = bv;
}

// 3-NN: strict-less insertion => earlier index wins on ties (top_k semantics).
__global__ void k_knn3(const float* __restrict__ x1, int N1,
                       const float* __restrict__ x2, int N2, int Bb,
                       int* __restrict__ idx, float* __restrict__ wgt) {
  int BN1 = Bb * N1, BN2 = Bb * N2;
  int m = GTID;
  if (m >= BN1) return;
  int b = m / N1;
  float ax = x1[m], ay = x1[BN1 + m], az = x1[2 * BN1 + m];
  float d0 = 3.4e38f, d1 = 3.4e38f, d2 = 3.4e38f;
  int i0 = b * N2, i1 = b * N2, i2 = b * N2;
  for (int j = 0; j < N2; ++j) {
    int p = b * N2 + j;
    float dx = x2[p] - ax, dy = x2[BN2 + p] - ay, dz = x2[2 * BN2 + p] - az;
    float d = dx * dx + dy * dy + dz * dz;
    if (d < d0)      { d2 = d1; i2 = i1; d1 = d0; i1 = i0; d0 = d; i0 = p; }
    else if (d < d1) { d2 = d1; i2 = i1; d1 = d;  i1 = p; }
    else if (d < d2) { d2 = d;  i2 = p; }
  }
  float w0 = 1.f / (d0 + 1e-8f), w1 = 1.f / (d1 + 1e-8f), w2 = 1.f / (d2 + 1e-8f);
  float sw = w0 + w1 + w2;
  idx[m * 3] = i0; idx[m * 3 + 1] = i1; idx[m * 3 + 2] = i2;
  wgt[m * 3] = w0 / sw; wgt[m * 3 + 1] = w1 / sw; wgt[m * 3 + 2] = w2 / sw;
}

// FP input: X = [P1 (C1), interp(P2) (C2), 0pad] packed f16
__global__ void k_interp(const float* __restrict__ P1, int C1, int s1,
                         const float* __restrict__ P2, int C2, int s2,
                         const int* __restrict__ idx, const float* __restrict__ wgt,
                         _Float16* __restrict__ X, int Cip) {
  int m = GTID;
  if (m >= s1) return;
  int i0 = idx[m * 3], i1 = idx[m * 3 + 1], i2 = idx[m * 3 + 2];
  float w0 = wgt[m * 3], w1 = wgt[m * 3 + 1], w2 = wgt[m * 3 + 2];
  for (int c = 0; c < C1; ++c)
    X[xidx(c, m, Cip)] = (_Float16)P1[(size_t)c * s1 + m];
  for (int c = 0; c < C2; ++c) {
    const float* r = P2 + (size_t)c * s2;
    X[xidx(C1 + c, m, Cip)] = (_Float16)(w0 * r[i0] + w1 * r[i1] + w2 * r[i2]);
  }
  for (int c = C1 + C2; c < Cip; ++c) X[xidx(c, m, Cip)] = (_Float16)0.f;
}

// flatten_sa input: [xyz(3), l3p(512), 0pad] packed, M = B*256
__global__ void k_flat(const float* __restrict__ nx3, const float* __restrict__ l3p,
                       _Float16* __restrict__ X, int Cip, int M) {
  int m = GTID;
  if (m >= M) return;
  for (int c = 0; c < 3; ++c) X[xidx(c, m, Cip)] = (_Float16)nx3[(size_t)c * M + m];
  for (int c = 0; c < 512; ++c)
    X[xidx(3 + c, m, Cip)] = (_Float16)l3p[(size_t)c * M + m];
  for (int c = 515; c < Cip; ++c) X[xidx(c, m, Cip)] = (_Float16)0.f;
}

__global__ void k_idppool(const float* __restrict__ Y, int Np, int Bb, int C,
                          float* __restrict__ out) {
  int id = GTID;
  if (id >= Bb * C) return;
  int b = id / C, c = id - b * C;
  int M = Bb * Np;
  const float* r = Y + (size_t)c * M + b * Np;
  float v = r[0];
  for (int i = 1; i < Np; ++i) v = fmaxf(v, r[i]);
  out[b * C + c] = v;
}

// fc1 -> layernorm -> relu -> fc2, one block per batch
__global__ void k_fchead(const float* __restrict__ idp, const float* __restrict__ w1,
                         const float* __restrict__ b1, const float* __restrict__ lng,
                         const float* __restrict__ lnb, const float* __restrict__ w2,
                         const float* __restrict__ b2, float* __restrict__ o) {
  __shared__ float r1[256], r2[256], hs[256];
  int b = blockIdx.x, t = threadIdx.x;
  const float* v = idp + b * 512;
  float acc = b1[t];
  const float* wr = w1 + (size_t)t * 512;
  for (int k = 0; k < 512; ++k) acc += wr[k] * v[k];
  r1[t] = acc; r2[t] = acc * acc;
  __syncthreads();
  for (int s = 128; s > 0; s >>= 1) {
    if (t < s) { r1[t] += r1[t + s]; r2[t] += r2[t + s]; }
    __syncthreads();
  }
  float mu = r1[0] * (1.f / 256.f);
  float var = r2[0] * (1.f / 256.f) - mu * mu;
  float h = (acc - mu) * rsqrtf(var + 1e-5f) * lng[t] + lnb[t];
  h = h > 0.f ? h : 0.f;
  hs[t] = h;
  __syncthreads();
  if (t < 8) {
    float a = b2[t];
    const float* w = w2 + t * 256;
    for (int k = 0; k < 256; ++k) a += w[k] * hs[k];
    o[b * 8 + t] = a;
  }
}

// pd1 = softmax over 2 channels; also feeds l0in2 rows 36/37
__global__ void k_pd1(const float* __restrict__ P, int S, int N,
                      const float* __restrict__ w, const float* __restrict__ bia,
                      float* __restrict__ outPd, float* __restrict__ l0in2) {
  int m = GTID;
  if (m >= S) return;
  int b = m / N, n = m - b * N;
  float a0 = bia[0], a1 = bia[1];
  for (int c = 0; c < 32; ++c) {
    float x = P[(size_t)c * S + m];
    a0 += w[c] * x;
    a1 += w[32 + c] * x;
  }
  float mx = fmaxf(a0, a1);
  float e0 = expf(a0 - mx), e1 = expf(a1 - mx);
  float inv = 1.f / (e0 + e1);
  float p0 = e0 * inv, p1 = e1 * inv;
  outPd[(size_t)b * 2 * N + n] = p0;
  outPd[(size_t)b * 2 * N + N + n] = p1;
  l0in2[(size_t)36 * S + m] = p0;
  l0in2[(size_t)37 * S + m] = p1;
}

// single-output 1x1 conv over 32 channels
__global__ void k_mask1(const float* __restrict__ P, int S,
                        const float* __restrict__ w, const float* __restrict__ bia,
                        float* __restrict__ out) {
  int m = GTID;
  if (m >= S) return;
  float a = bia[0];
  for (int c = 0; c < 32; ++c) a += w[c] * P[(size_t)c * S + m];
  out[m] = a;
}

// ---------------------------------------------------------------------------
// Host orchestration
// ---------------------------------------------------------------------------
struct ConvP { const float *w, *b, *g, *bt, *mn, *vr; int cin, cout; };

static ConvP getconv(void* const* din, int base, int cin, int cout) {
  // bn conv leaves sorted: b, beta, gamma, mean, var, w
  ConvP c;
  c.b  = (const float*)din[base + 0];
  c.bt = (const float*)din[base + 1];
  c.g  = (const float*)din[base + 2];
  c.mn = (const float*)din[base + 3];
  c.vr = (const float*)din[base + 4];
  c.w  = (const float*)din[base + 5];
  c.cin = cin; c.cout = cout;
  return c;
}

static void run_cbr(hipStream_t st, const ConvP& c, const _Float16* X, int M,
                    float* Yf, _Float16* Yh, _Float16* Wh, float* sb, float* tb) {
  int Cip = pad32i(c.cin), Cop = pad32i(c.cout);
  int tot = Cop * Cip;
  k_pack<<<ceil_div(tot, 256), 256, 0, st>>>(c.w, c.b, c.g, c.bt, c.mn, c.vr,
                                             c.cout, c.cin, Cop, Cip, Wh, sb, tb);
  int Mtiles = ceil_div(M, 16);               // always a multiple of 4 here
  dim3 g(Cop / 16, ceil_div(ceil_div(Mtiles, 4), 4));
  k_wmma_gemm<<<g, 128, 0, st>>>(Wh, sb, tb, X, Yf, Yh, M, Cip, Cop, Mtiles);
}

extern "C" void kernel_launch(void* const* d_in, const int* in_sizes, int n_in,
                              void* d_out, int out_size, void* d_ws, size_t ws_size,
                              hipStream_t stream) {
  (void)in_sizes; (void)n_in; (void)out_size; (void)ws_size;
  const int B = 2, N = 16000, BN0 = B * N;
  const float* xyz = (const float*)d_in[0];
  float* out = (float*)d_out;

  // ---- workspace bump allocation ----
  size_t off = 0;
  char* base = (char*)d_ws;
  auto alloc = [&](size_t bytes) -> void* {
    off = (off + 255) & ~(size_t)255;
    void* p = base + off;
    off += bytes;
    return p;
  };
#define AF(n) ((float*)alloc((size_t)(n) * sizeof(float)))
#define AH(n) ((_Float16*)alloc((size_t)(n) * sizeof(_Float16)))
#define AI(n) ((int*)alloc((size_t)(n) * sizeof(int)))

  float* pts0   = AF(36 * BN0);       // CM copy of xyz (rows 0..2 = coords)
  float* l0in2  = AF(38 * BN0);       // pass-2 input features
  float* distws = AF(BN0);
  int* fidx1 = AI(B * 1024); int* fidx2 = AI(B * 512); int* fidx3 = AI(B * 256);
  float* nx1 = AF(3 * B * 1024); float* nx2 = AF(3 * B * 512); float* nx3 = AF(3 * B * 256);
  int* g1a = AI(B * 1024 * 16); int* g1b = AI(B * 1024 * 32);
  int* g2a = AI(B * 512 * 16);  int* g2b = AI(B * 512 * 32);
  int* g3a = AI(B * 256 * 16);  int* g3b = AI(B * 256 * 32);
  int* idx3  = AI(B * 512 * 3);   float* w3  = AF(B * 512 * 3);
  int* idxf2 = AI(B * 1024 * 3);  float* wf2 = AF(B * 1024 * 3);
  int* idxf1 = AI(B * 16000 * 3); float* wf1 = AF(B * 16000 * 3);
  _Float16* Xg  = AH(6400000);
  float* Y1f    = AF(3800000);
  _Float16* Y1h = AH(3800000);
  float* Y2f    = AF(4300000);
  float* l1p  = AF(64 * 2048);  float* l2p  = AF(256 * 1024); float* l3p = AF(512 * 512);
  float* l2pn = AF(256 * 1024); float* l1pn = AF(128 * 2048); float* l0p = AF(32 * BN0);
  float* sidp = AF(B * 512);
  _Float16* Wh = AH(160000);
  float* sb = AF(512); float* tb = AF(512);

  // ---- geometry (shared by both passes: coordinates identical) ----
  k_bmaj2cm<<<ceil_div(36 * BN0, 256), 256, 0, stream>>>(xyz, B, 36, N, pts0);
  hipMemcpyAsync(l0in2, pts0, (size_t)36 * BN0 * sizeof(float),
                 hipMemcpyDeviceToDevice, stream);

  k_fps<<<B, 256, 0, stream>>>(pts0, BN0, N, 1024, distws, fidx1);
  k_centers<<<ceil_div(B * 1024, 256), 256, 0, stream>>>(pts0, BN0, fidx1, N, 1024, B, nx1);
  k_fps<<<B, 256, 0, stream>>>(nx1, B * 1024, 1024, 512, distws, fidx2);
  k_centers<<<ceil_div(B * 512, 256), 256, 0, stream>>>(nx1, B * 1024, fidx2, 1024, 512, B, nx2);
  k_fps<<<B, 256, 0, stream>>>(nx2, B * 512, 512, 256, distws, fidx3);
  k_centers<<<ceil_div(B * 256, 256), 256, 0, stream>>>(nx2, B * 512, fidx3, 512, 256, B, nx3);

  k_ball<<<ceil_div(B * 1024, 128), 128, 0, stream>>>(pts0, BN0, N, nx1, 1024, B, 0.0025f * 0.0025f, 16, g1a);
  k_ball<<<ceil_div(B * 1024, 128), 128, 0, stream>>>(pts0, BN0, N, nx1, 1024, B, 0.005f * 0.005f, 32, g1b);
  k_ball<<<ceil_div(B * 512, 128), 128, 0, stream>>>(nx1, B * 1024, 1024, nx2, 512, B, 0.005f * 0.005f, 16, g2a);
  k_ball<<<ceil_div(B * 512, 128), 128, 0, stream>>>(nx1, B * 1024, 1024, nx2, 512, B, 0.01f * 0.01f, 32, g2b);
  k_ball<<<ceil_div(B * 256, 128), 128, 0, stream>>>(nx2, B * 512, 512, nx3, 256, B, 0.01f * 0.01f, 16, g3a);
  k_ball<<<ceil_div(B * 256, 128), 128, 0, stream>>>(nx2, B * 512, 512, nx3, 256, B, 0.02f * 0.02f, 32, g3b);

  k_knn3<<<ceil_div(B * 512, 256), 256, 0, stream>>>(nx2, 512, nx3, 256, B, idx3, w3);
  k_knn3<<<ceil_div(B * 1024, 256), 256, 0, stream>>>(nx1, 1024, nx2, 512, B, idxf2, wf2);
  k_knn3<<<ceil_div(BN0, 256), 256, 0, stream>>>(pts0, N, nx1, 1024, B, idxf1, wf1);

  // param base indices (d_in[0]=xyz; params flattened with sorted dict keys)
  const int saBase[3][2] = {{95, 119}, {143, 167}, {191, 215}};
  const int fpBase[3][2] = {{65, 77}, {41, 53}, {17, 29}};  // fp3, fp2, fp1

  for (int pass = 0; pass < 2; ++pass) {
    const float* P0 = pass ? l0in2 : pts0;
    int Cf0 = pass ? 38 : 36;

    // ---- SA chain (3 levels x 2 radius branches) ----
    const float* feat = P0; int featStride = BN0; int Cf = Cf0;
    const float* srcc[3] = {pts0, nx1, nx2};
    const int srcStride[3] = {BN0, B * 1024, B * 512};
    const float* ctrs[3] = {nx1, nx2, nx3};
    const int Nss[3] = {N, 1024, 512}, Nps[3] = {1024, 512, 256};
    int* gI[3][2] = {{g1a, g1b}, {g2a, g2b}, {g3a, g3b}};
    float* dstP[3] = {l1p, l2p, l3p};
    const int co0[3] = {32, 64, 196}, co1[3] = {32, 128, 256};

    for (int lvl = 0; lvl < 3; ++lvl) {
      int Np = Nps[lvl], Ns = Nss[lvl], BNq = B * Np;
      for (int br = 0; br < 2; ++br) {
        int K = br ? 32 : 16;
        int M = BNq * K;
        int Cin = Cf + 3, Cip = pad32i(Cin);
        k_group<<<ceil_div(M, 256), 256, 0, stream>>>(
            feat, Cf, featStride, srcc[lvl], srcStride[lvl], ctrs[lvl],
            gI[lvl][br], Ns, Np, K, Xg, Cip, M, BNq);
        ConvP c0 = getconv(d_in, saBase[lvl][pass] + br * 12, Cin, co0[lvl]);
        ConvP c1 = getconv(d_in, saBase[lvl][pass] + br * 12 + 6, co0[lvl], co1[lvl]);
        run_cbr(stream, c0, Xg, M, Y1f, Y1h, Wh, sb, tb);
        run_cbr(stream, c1, Y1h, M, Y2f, nullptr, Wh, sb, tb);
        k_maxpool<<<ceil_div(BNq * co1[lvl], 256), 256, 0, stream>>>(
            Y2f, M, co1[lvl], K, BNq, dstP[lvl], co1[lvl] * br);
      }
      feat = dstP[lvl]; featStride = BNq; Cf = 2 * co1[lvl];
    }

    // ---- FP chain ----
    k_interp<<<ceil_div(B * 512, 256), 256, 0, stream>>>(
        l2p, 256, B * 512, l3p, 512, B * 256, idx3, w3, Xg, 768);
    {
      ConvP c0 = getconv(d_in, fpBase[0][pass], 768, 256);
      ConvP c1 = getconv(d_in, fpBase[0][pass] + 6, 256, 256);
      run_cbr(stream, c0, Xg, B * 512, Y1f, Y1h, Wh, sb, tb);
      run_cbr(stream, c1, Y1h, B * 512, l2pn, nullptr, Wh, sb, tb);
    }
    k_interp<<<ceil_div(B * 1024, 256), 256, 0, stream>>>(
        l1p, 64, B * 1024, l2pn, 256, B * 512, idxf2, wf2, Xg, 320);
    {
      ConvP c0 = getconv(d_in, fpBase[1][pass], 320, 128);
      ConvP c1 = getconv(d_in, fpBase[1][pass] + 6, 128, 128);
      run_cbr(stream, c0, Xg, B * 1024, Y1f, Y1h, Wh, sb, tb);
      run_cbr(stream, c1, Y1h, B * 1024, l1pn, nullptr, Wh, sb, tb);
    }
    int Cin1 = Cf0 + 128, Cip1 = pad32i(Cin1);
    k_interp<<<ceil_div(BN0, 256), 256, 0, stream>>>(
        P0, Cf0, BN0, l1pn, 128, B * 1024, idxf1, wf1, Xg, Cip1);
    {
      ConvP c0 = getconv(d_in, fpBase[2][pass], Cin1, 64);
      ConvP c1 = getconv(d_in, fpBase[2][pass] + 6, 64, 32);
      run_cbr(stream, c0, Xg, BN0, Y1f, Y1h, Wh, sb, tb);
      run_cbr(stream, c1, Y1h, BN0, l0p, nullptr, Wh, sb, tb);
    }

    if (pass == 0) {
      k_pd1<<<ceil_div(BN0, 256), 256, 0, stream>>>(
          l0p, BN0, N, (const float*)d_in[92], (const float*)d_in[91], out, l0in2);
      k_mask1<<<ceil_div(BN0, 256), 256, 0, stream>>>(
          l0p, BN0, (const float*)d_in[240], (const float*)d_in[239], out + 64000);
    } else {
      k_mask1<<<ceil_div(BN0, 256), 256, 0, stream>>>(
          l0p, BN0, (const float*)d_in[94], (const float*)d_in[93], out + 96000);
    }
  }

  // ---- flatten_sa on pass-2 l3 features + fc head ----
  k_flat<<<ceil_div(B * 256, 256), 256, 0, stream>>>(nx3, l3p, Xg, 544, B * 256);
  {
    ConvP c0 = getconv(d_in, 5, 515, 256);
    ConvP c1 = getconv(d_in, 11, 256, 512);
    run_cbr(stream, c0, Xg, B * 256, Y1f, Y1h, Wh, sb, tb);
    run_cbr(stream, c1, Y1h, B * 256, Y2f, nullptr, Wh, sb, tb);
  }
  k_idppool<<<ceil_div(B * 512, 256), 256, 0, stream>>>(Y2f, 256, B, 512, sidp);
  k_fchead<<<B, 256, 0, stream>>>(sidp, (const float*)d_in[2], (const float*)d_in[1],
                                  (const float*)d_in[90], (const float*)d_in[89],
                                  (const float*)d_in[4], (const float*)d_in[3],
                                  out + 128000);
#undef AF
#undef AH
#undef AI
}
